// SelfAttention_40638980555138
// MI455X (gfx1250) — matrix operand order repaired and checked by reading, not compiled, not run on hardware
//
#include <hip/hip_runtime.h>
#include <hip/hip_bf16.h>
#include <hip/hip_fp16.h>

typedef __attribute__((ext_vector_type(16))) _Float16 v16h;
typedef __attribute__((ext_vector_type(8)))  _Float16 v8h;
typedef __attribute__((ext_vector_type(8)))  float    v8f;

#define WMMA_F16(A, B, Cacc) \
  __builtin_amdgcn_wmma_f32_16x16x32_f16(false, (A), false, (B), (short)0, (Cacc), false, false)

// ---------------------------------------------------------------------------
// WMMA operand loaders (wave32).
// A 16x32 f16: lanes 0-15 -> M=lane, K 0..7 then 16..23; lanes 16-31 -> K 8..15, 24..31
// B 32x16 f16 (source stored [n][k] row-major): lanes 0-15 -> N=lane, K 0..7 then 16..23;
//                                               lanes 16-31 -> N=lane-16, K 8..15, 24..31
// ---------------------------------------------------------------------------
__device__ __forceinline__ v16h load_a16x32(const _Float16* tile, int ld, int lane) {
  const _Float16* p = tile + (size_t)(lane & 15) * ld + ((lane & 16) ? 8 : 0);
  v8h lo = *(const v8h*)(p);
  v8h hi = *(const v8h*)(p + 16);
  v16h r;
#pragma unroll
  for (int i = 0; i < 8; ++i) { r[i] = lo[i]; r[i + 8] = hi[i]; }
  return r;
}

__device__ __forceinline__ v16h load_b32x16(const _Float16* tile, int ld, int lane) {
  const _Float16* p = tile + (size_t)(lane & 15) * ld + ((lane & 16) ? 8 : 0);
  v8h lo = *(const v8h*)(p);
  v8h hi = *(const v8h*)(p + 16);
  v16h r;
#pragma unroll
  for (int i = 0; i < 8; ++i) { r[i] = lo[i]; r[i + 8] = hi[i]; }
  return r;
}

// Async 16-byte-per-lane copy global -> LDS (CDNA5, tracked by ASYNCcnt).
__device__ __forceinline__ void async_copy16(unsigned int lds_off, const void* g) {
  asm volatile("global_load_async_to_lds_b128 %0, %1, off"
               :: "v"(lds_off), "v"(g) : "memory");
}

// ---------------------------------------------------------------------------
// Kernel 1: weight transpose + fp16 convert.  WT[n][k] = (f16) W[k][n]
// ---------------------------------------------------------------------------
__global__ void wconv_kernel(const float* __restrict__ W, _Float16* __restrict__ WT) {
  int n = blockIdx.x;      // 256
  int k = threadIdx.x;     // 256
  WT[n * 256 + k] = (_Float16)W[k * 256 + n];
}

// ---------------------------------------------------------------------------
// Kernel 2: GroupNorm -> fp16.  One block per (batch, group); 32 groups of 8ch.
// ---------------------------------------------------------------------------
__global__ void gn_kernel(const float* __restrict__ x, const float* __restrict__ scale,
                          const float* __restrict__ bias, _Float16* __restrict__ ft) {
  const int bg = blockIdx.x;            // b*32 + g
  const int b = bg >> 5, g = bg & 31;
  const int t = threadIdx.x;            // 256 threads
  const float* xb = x + (size_t)b * 1024 * 256;

  float s = 0.f, s2 = 0.f;
#pragma unroll
  for (int i = 0; i < 32; ++i) {
    int idx = t + i * 256;              // 0..8191 over 1024 positions x 8 channels
    int p = idx >> 3, ch = g * 8 + (idx & 7);
    float v = xb[p * 256 + ch];
    s += v; s2 += v * v;
  }
  __shared__ float red0[256];
  __shared__ float red1[256];
  red0[t] = s; red1[t] = s2;
  __syncthreads();
  for (int off = 128; off > 0; off >>= 1) {
    if (t < off) { red0[t] += red0[t + off]; red1[t] += red1[t + off]; }
    __syncthreads();
  }
  const float mean = red0[0] * (1.f / 8192.f);
  const float var  = red1[0] * (1.f / 8192.f) - mean * mean;
  const float rstd = rsqrtf(var + 1e-5f);

  _Float16* ftb = ft + (size_t)b * 1024 * 256;
#pragma unroll
  for (int i = 0; i < 32; ++i) {
    int idx = t + i * 256;
    int p = idx >> 3, ch = g * 8 + (idx & 7);
    float v = (xb[p * 256 + ch] - mean) * rstd * scale[ch] + bias[ch];
    ftb[p * 256 + ch] = (_Float16)v;
  }
}

// ---------------------------------------------------------------------------
// Kernel 3: QKV projections.  grid=(2048, 3), block=128 (4 waves).
// Each wave: 16x64 output tile, K loop over 256 in steps of 32.
// q gets 1/sqrt(C) folded in; v is stored transposed [b][c][token].
// ---------------------------------------------------------------------------
__global__ void qkv_kernel(const _Float16* __restrict__ ft,
                           const _Float16* __restrict__ WTq, const _Float16* __restrict__ WTk,
                           const _Float16* __restrict__ WTv,
                           const float* __restrict__ bq, const float* __restrict__ bk,
                           const float* __restrict__ bv,
                           _Float16* __restrict__ qb, _Float16* __restrict__ kb,
                           _Float16* __restrict__ vT) {
  const int lane  = threadIdx.x & 31;
  const int wave  = threadIdx.x >> 5;   // 0..3
  const int m0    = blockIdx.x * 16;    // token-tile base (0..32767)
  const int n0    = wave * 64;          // output-channel base
  const int which = blockIdx.y;         // 0:q 1:k 2:v

  const _Float16* WT  = (which == 0) ? WTq : (which == 1) ? WTk : WTv;
  const float*    bia = (which == 0) ? bq  : (which == 1) ? bk  : bv;

  v8f acc[4] = {};
  for (int kk = 0; kk < 256; kk += 32) {
    v16h a = load_a16x32(ft + (size_t)m0 * 256 + kk, 256, lane);
#pragma unroll
    for (int nt = 0; nt < 4; ++nt) {
      v16h b = load_b32x16(WT + (size_t)(n0 + nt * 16) * 256 + kk, 256, lane);
      acc[nt] = WMMA_F16(a, b, acc[nt]);
    }
  }

  const int col   = lane & 15;
  const int rbase = (lane & 16) ? 8 : 0;

  if (which == 2) {
    // v: store transposed [b][c][token]; per-lane 8 consecutive tokens -> one b128 store
    const int b  = m0 >> 10;
    const int sp = m0 & 1023;
#pragma unroll
    for (int nt = 0; nt < 4; ++nt) {
      const int c = n0 + nt * 16 + col;
      const float bb = bia[c];
      v8h packed;
#pragma unroll
      for (int r = 0; r < 8; ++r) packed[r] = (_Float16)(acc[nt][r] + bb);
      *(v8h*)(vT + ((size_t)b * 256 + c) * 1024 + sp + rbase) = packed;
    }
  } else {
    _Float16* dst = (which == 0) ? qb : kb;
    const float sc = (which == 0) ? 0.0625f : 1.f;   // 1/sqrt(256) folded into q
#pragma unroll
    for (int nt = 0; nt < 4; ++nt) {
      const int c = n0 + nt * 16 + col;
      const float bb = bia[c];
#pragma unroll
      for (int r = 0; r < 8; ++r)
        dst[(size_t)(m0 + rbase + r) * 256 + c] = (_Float16)((acc[nt][r] + bb) * sc);
    }
  }
}

// ---------------------------------------------------------------------------
// Kernel 4: flash attention with double-buffered async K/V staging in LDS.
// block = 128 threads (4 waves), grid = 512.  One wave per 16-query tile;
// the 4 waves of a block share the same batch and therefore the same K/V
// tiles, which are staged cooperatively with global_load_async_to_lds_b128.
// ---------------------------------------------------------------------------
#define KPAD 264   // K-tile LDS row pitch in halves (528B: 16-lane bank-conflict free)
#define VPAD 40    // V-tile LDS row pitch in halves (80B:  16-lane bank-conflict free)

__global__ void attn_kernel(const _Float16* __restrict__ qbuf,
                            const _Float16* __restrict__ kbuf,
                            const _Float16* __restrict__ vT,
                            _Float16* __restrict__ obuf) {
  const int lane = threadIdx.x & 31;
  const int wave = threadIdx.x >> 5;
  const int tid  = threadIdx.x;             // 0..127
  const int qt   = blockIdx.x * 4 + wave;   // 0..2047
  const int b    = qt >> 6;                 // 64 q-tiles per batch (4 waves share b)
  const int m0   = (qt & 63) * 16;

  const _Float16* qb = qbuf + (size_t)b * 1024 * 256;
  const _Float16* kb = kbuf + (size_t)b * 1024 * 256;
  const _Float16* vb = vT   + (size_t)b * 256 * 1024;

  __shared__ __align__(16) _Float16 ksh[2][32 * KPAD];    // [key][c], padded rows
  __shared__ __align__(16) _Float16 vsh[2][256 * VPAD];   // [c][key], padded rows
  __shared__ __align__(16) _Float16 pshare[4][16 * 32];   // wave-private P staging
  _Float16* pl = pshare[wave];

  // ---- async stage of one 32-key K tile (16KB) + V tile (16KB) into buffer bf
  auto stage = [&](int kt, int bf) {
    const char* kg = (const char*)(kb + (size_t)kt * 256);     // 32 rows x 512B contiguous
    unsigned int klds = (unsigned int)(uintptr_t)(&ksh[bf][0]);
#pragma unroll
    for (int j = 0; j < 8; ++j) {
      int idx = tid + j * 128;            // 0..1023 chunks of 16B
      int row = idx >> 5, sub = idx & 31;
      async_copy16(klds + (unsigned int)(row * (KPAD * 2) + sub * 16),
                   kg + (size_t)row * 512 + sub * 16);
    }
    const char* vg = (const char*)vb + (size_t)kt * 2;         // 256 rows x 64B, pitch 2048B
    unsigned int vlds = (unsigned int)(uintptr_t)(&vsh[bf][0]);
#pragma unroll
    for (int j = 0; j < 8; ++j) {
      int idx = tid + j * 128;            // 0..1023 chunks of 16B
      int row = idx >> 2, sub = idx & 3;
      async_copy16(vlds + (unsigned int)(row * (VPAD * 2) + sub * 16),
                   vg + (size_t)row * 2048 + sub * 16);
    }
  };

  stage(0, 0);   // prefetch first tile

  // Q tile resident in registers: 8 chunks of 16x32 fp16 (64 VGPRs)
  v16h Q[8];
#pragma unroll
  for (int c = 0; c < 8; ++c)
    Q[c] = load_a16x32(qb + (size_t)m0 * 256 + c * 32, 256, lane);

  v8f O[16];
#pragma unroll
  for (int t = 0; t < 16; ++t) O[t] = (v8f)(0.f);
  float mrow[8], lrow[8];
#pragma unroll
  for (int r = 0; r < 8; ++r) { mrow[r] = -1e30f; lrow[r] = 0.f; }

  const int col    = lane & 15;
  const int rowoff = (lane & 16) ? 8 : 0;

  for (int it = 0; it < 32; ++it) {       // 32 steps of 32 keys
    const int bf = it & 1;
    if (it + 1 < 32) {
      stage((it + 1) * 32, bf ^ 1);       // prefetch next tile (16 async/wave in flight)
      asm volatile("s_wait_asynccnt 16" ::: "memory");   // current tile landed
    } else {
      asm volatile("s_wait_asynccnt 0" ::: "memory");
    }
    __syncthreads();                      // all waves' copies complete

    const _Float16* K0 = &ksh[bf][0];
    const _Float16* V0 = &vsh[bf][0];

    // S = Q * K^T  (16 x 32), accumulate over C=256, operands from LDS
    v8f S0 = (v8f)(0.f), S1 = (v8f)(0.f);
#pragma unroll
    for (int c = 0; c < 8; ++c) {
      v16h B0 = load_b32x16(K0 + c * 32,             KPAD, lane);
      v16h B1 = load_b32x16(K0 + 16 * KPAD + c * 32, KPAD, lane);
      S0 = WMMA_F16(Q[c], B0, S0);
      S1 = WMMA_F16(Q[c], B1, S1);
    }

    // online softmax: row stats via 16-lane xor-shuffle reduction
    float alpha[8];
#pragma unroll
    for (int r = 0; r < 8; ++r) {
      float mx = fmaxf(S0[r], S1[r]);
#pragma unroll
      for (int off = 1; off < 16; off <<= 1) mx = fmaxf(mx, __shfl_xor(mx, off, 32));
      float mnew = fmaxf(mrow[r], mx);
      alpha[r] = __expf(mrow[r] - mnew);
      mrow[r] = mnew;
      float p0 = __expf(S0[r] - mnew);
      float p1 = __expf(S1[r] - mnew);
      float rs = p0 + p1;
#pragma unroll
      for (int off = 1; off < 16; off <<= 1) rs += __shfl_xor(rs, off, 32);
      lrow[r] = lrow[r] * alpha[r] + rs;
      // stage P (C-layout -> LDS, read back in A-layout)
      pl[(r + rowoff) * 32 + col]      = (_Float16)p0;
      pl[(r + rowoff) * 32 + col + 16] = (_Float16)p1;
    }

    // rescale running accumulator
#pragma unroll
    for (int t = 0; t < 16; ++t)
#pragma unroll
      for (int r = 0; r < 8; ++r) O[t][r] *= alpha[r];

    asm volatile("s_wait_dscnt 0" ::: "memory");
    v16h P = load_a16x32(pl, 32, lane);

    // O += P * V   (V tile from LDS [c][key])
#pragma unroll
    for (int t = 0; t < 16; ++t) {
      v16h Bv = load_b32x16(V0 + (size_t)(t * 16) * VPAD, VPAD, lane);
      O[t] = WMMA_F16(P, Bv, O[t]);
    }

    __syncthreads();                      // reads done before buffer is overwritten
  }

  // normalize and store fp16 [b][token][c]
#pragma unroll
  for (int r = 0; r < 8; ++r) lrow[r] = 1.f / lrow[r];
  _Float16* ob = obuf + (size_t)b * 1024 * 256;
#pragma unroll
  for (int t = 0; t < 16; ++t) {
    const int c = t * 16 + col;
#pragma unroll
    for (int r = 0; r < 8; ++r)
      ob[(size_t)(m0 + rowoff + r) * 256 + c] = (_Float16)(O[t][r] * lrow[r]);
  }
}

// ---------------------------------------------------------------------------
// Kernel 5: output projection + bias + residual (f32 out).
// grid = 2048, block = 128 (4 waves, wave = 16x64 tile).
// ---------------------------------------------------------------------------
__global__ void proj_kernel(const _Float16* __restrict__ obuf,
                            const _Float16* __restrict__ WTo,
                            const float* __restrict__ bo,
                            const float* __restrict__ x,
                            float* __restrict__ out) {
  const int lane = threadIdx.x & 31;
  const int wave = threadIdx.x >> 5;
  const int m0   = blockIdx.x * 16;
  const int n0   = wave * 64;

  v8f acc[4] = {};
  for (int kk = 0; kk < 256; kk += 32) {
    v16h a = load_a16x32(obuf + (size_t)m0 * 256 + kk, 256, lane);
#pragma unroll
    for (int nt = 0; nt < 4; ++nt) {
      v16h b = load_b32x16(WTo + (size_t)(n0 + nt * 16) * 256 + kk, 256, lane);
      acc[nt] = WMMA_F16(a, b, acc[nt]);
    }
  }

  const int col   = lane & 15;
  const int rbase = (lane & 16) ? 8 : 0;
#pragma unroll
  for (int nt = 0; nt < 4; ++nt) {
    const int c = n0 + nt * 16 + col;
    const float bb = bo[c];
#pragma unroll
    for (int r = 0; r < 8; ++r) {
      const size_t idx = (size_t)(m0 + rbase + r) * 256 + c;
      out[idx] = acc[nt][r] + bb + x[idx];
    }
  }
}

// ---------------------------------------------------------------------------
extern "C" void kernel_launch(void* const* d_in, const int* in_sizes, int n_in,
                              void* d_out, int out_size, void* d_ws, size_t ws_size,
                              hipStream_t stream) {
  const float* x        = (const float*)d_in[0];
  const float* gn_scale = (const float*)d_in[1];
  const float* gn_bias  = (const float*)d_in[2];
  const float* Wq = (const float*)d_in[3]; const float* bq = (const float*)d_in[4];
  const float* Wk = (const float*)d_in[5]; const float* bk = (const float*)d_in[6];
  const float* Wv = (const float*)d_in[7]; const float* bv = (const float*)d_in[8];
  const float* Wo = (const float*)d_in[9]; const float* bo = (const float*)d_in[10];
  float* out = (float*)d_out;

  const size_t NTOK = 32 * 1024;   // B * H * W
  const size_t CH   = 256;

  _Float16* ft  = (_Float16*)d_ws;        // normalized features, fp16 [b][tok][c]
  _Float16* qb  = ft  + NTOK * CH;        // q (pre-scaled)      [b][tok][c]
  _Float16* kb  = qb  + NTOK * CH;        // k                   [b][tok][c]
  _Float16* vTb = kb  + NTOK * CH;        // v transposed        [b][c][tok]
  _Float16* ob  = vTb + NTOK * CH;        // attention output    [b][tok][c]
  _Float16* WTq = ob  + NTOK * CH;        // weights, fp16, [n][k]
  _Float16* WTk = WTq + CH * CH;
  _Float16* WTv = WTk + CH * CH;
  _Float16* WTo = WTv + CH * CH;

  wconv_kernel<<<256, 256, 0, stream>>>(Wq, WTq);
  wconv_kernel<<<256, 256, 0, stream>>>(Wk, WTk);
  wconv_kernel<<<256, 256, 0, stream>>>(Wv, WTv);
  wconv_kernel<<<256, 256, 0, stream>>>(Wo, WTo);

  gn_kernel<<<1024, 256, 0, stream>>>(x, gn_scale, gn_bias, ft);

  qkv_kernel<<<dim3(2048, 3), 128, 0, stream>>>(ft, WTq, WTk, WTv, bq, bk, bv, qb, kb, vTb);

  attn_kernel<<<512, 128, 0, stream>>>(qb, kb, vTb, ob);

  proj_kernel<<<2048, 128, 0, stream>>>(ob, WTo, bo, x, out);
}
